// QuantizedLinear_26731876451110
// MI455X (gfx1250) — matrix operand order repaired and checked by reading, not compiled, not run on hardware
//
#include <hip/hip_runtime.h>

// ---- problem constants (match reference) ----
#define IN_F   4096
#define OUT_F  16384
#define M_TOT  8192          // B * S = 4 * 2048
#define BM     128
#define BN     256
#define BK     32
#define NK     (IN_F / BK)   // 128 K-steps
#define LDK    40            // padded LDS row stride in f16 (80B, 16B multiple)

typedef _Float16 h4  __attribute__((ext_vector_type(4)));
typedef _Float16 h8  __attribute__((ext_vector_type(8)));
typedef _Float16 h16 __attribute__((ext_vector_type(16)));
typedef float    f4v __attribute__((ext_vector_type(4)));
typedef float    v8f __attribute__((ext_vector_type(8)));
typedef int      i4v __attribute__((ext_vector_type(4)));

__global__ __launch_bounds__(256)
void qlinear_wmma_f16(const float* __restrict__ x,
                      const int*   __restrict__ wq,
                      const float* __restrict__ wscale,
                      const float* __restrict__ wzp,
                      const float* __restrict__ bias,
                      float* __restrict__ out)
{
    __shared__ __align__(16) _Float16 As[2][BM * LDK];
    __shared__ __align__(16) _Float16 Bs[2][BN * LDK];

    const int t     = threadIdx.x;
    const int wave  = t >> 5;
    const int lane  = t & 31;
    const int lr    = lane & 15;
    const int half  = lane >> 4;
    const int waveM = wave >> 2;     // 0..1  -> 64-row band
    const int waveN = wave & 3;      // 0..3  -> 64-col band

    const int bm = blockIdx.x * BM;  // M block fastest-varying: weight band reused in L2
    const int bn = blockIdx.y * BN;

    // cooperative loader mapping: 8 threads per row, 4 elems each, 32 rows per pass
    const int lrow = t >> 3;         // 0..31
    const int lk   = (t & 7) * 4;    // 0..28

    // per-output-channel dequant params: fixed for the whole K-loop of this block
    // w = (q - zp)*s  ==  fma(q, s, -zp*s)
    float sc[8], nzs[8];
#pragma unroll
    for (int p = 0; p < 8; ++p) {
        const int o = bn + p * 32 + lrow;
        sc[p]  = wscale[o];
        nzs[p] = -wzp[o] * sc[p];
    }

    const float* xA = x  + (size_t)bm * IN_F;
    const int*   wB = wq + (size_t)bn * IN_F;

    f4v aReg[4];
    i4v bReg[8];

    // ---- prologue: load + convert tile 0 ----
#pragma unroll
    for (int p = 0; p < 4; ++p)
        aReg[p] = *(const f4v*)(xA + (size_t)(p * 32 + lrow) * IN_F + lk);
#pragma unroll
    for (int p = 0; p < 8; ++p)
        bReg[p] = *(const i4v*)(wB + (size_t)(p * 32 + lrow) * IN_F + lk);

#pragma unroll
    for (int p = 0; p < 4; ++p) {
        h4 av;
#pragma unroll
        for (int j = 0; j < 4; ++j) av[j] = (_Float16)aReg[p][j];
        *(h4*)(&As[0][(p * 32 + lrow) * LDK + lk]) = av;
    }
#pragma unroll
    for (int p = 0; p < 8; ++p) {
        h4 bv;
#pragma unroll
        for (int j = 0; j < 4; ++j)
            bv[j] = (_Float16)__builtin_fmaf((float)bReg[p][j], sc[p], nzs[p]);
        *(h4*)(&Bs[0][(p * 32 + lrow) * LDK + lk]) = bv;
    }
    __syncthreads();

    v8f acc[4][4];
#pragma unroll
    for (int fm = 0; fm < 4; ++fm)
#pragma unroll
        for (int fn = 0; fn < 4; ++fn)
            acc[fm][fn] = (v8f){0.f, 0.f, 0.f, 0.f, 0.f, 0.f, 0.f, 0.f};

    // ---- main K loop, double-buffered LDS ----
    for (int kt = 0; kt < NK; ++kt) {
        const int cur = kt & 1;

        // issue next-tile global loads (conversion deferred until after compute)
        if (kt + 1 < NK) {
            const int k0 = (kt + 1) * BK;
#pragma unroll
            for (int p = 0; p < 4; ++p)
                aReg[p] = *(const f4v*)(xA + (size_t)(p * 32 + lrow) * IN_F + k0 + lk);
#pragma unroll
            for (int p = 0; p < 8; ++p)
                bReg[p] = *(const i4v*)(wB + (size_t)(p * 32 + lrow) * IN_F + k0 + lk);
            if (kt + 4 < NK) {   // L2 prefetch hints a few tiles ahead
                __builtin_prefetch(xA + (size_t)lrow * IN_F + (kt + 4) * BK + lk, 0, 0);
                __builtin_prefetch(wB + (size_t)lrow * IN_F + (kt + 4) * BK + lk, 0, 0);
            }
        }

        // fragment loads from LDS (per ISA VGPR layouts)
        h16 af[4], bf[4];
#pragma unroll
        for (int fm = 0; fm < 4; ++fm) {
            const int row = waveM * 64 + fm * 16 + lr;
            const _Float16* pa = &As[cur][row * LDK + half * 8];
            h8 lo = *(const h8*)pa;          // K = h*8 .. h*8+7
            h8 hi = *(const h8*)(pa + 16);   // K = h*8+16 .. h*8+23
            af[fm] = __builtin_shufflevector(lo, hi,
                     0, 1, 2, 3, 4, 5, 6, 7, 8, 9, 10, 11, 12, 13, 14, 15);
        }
#pragma unroll
        for (int fn = 0; fn < 4; ++fn) {
            const int col = waveN * 64 + fn * 16 + lr;
            const _Float16* pb = &Bs[cur][col * LDK + half * 8];
            bf[fn] = __builtin_shufflevector(*(const h8*)pb, *(const h8*)(pb + 16),
                     0, 1, 2, 3, 4, 5, 6, 7, 8, 9, 10, 11, 12, 13, 14, 15); // K = h*8..+7, h*8+16..+23
        }

        // 16 WMMAs: 64x64 patch per wave
#pragma unroll
        for (int fm = 0; fm < 4; ++fm)
#pragma unroll
            for (int fn = 0; fn < 4; ++fn)
                acc[fm][fn] = __builtin_amdgcn_wmma_f32_16x16x32_f16(
                    false, af[fm], false, bf[fn],
                    (short)0, acc[fm][fn], false, false);

        // convert + stage next tile into the other LDS buffer
        if (kt + 1 < NK) {
            const int nxt = cur ^ 1;
            __syncthreads();
#pragma unroll
            for (int p = 0; p < 4; ++p) {
                h4 av;
#pragma unroll
                for (int j = 0; j < 4; ++j) av[j] = (_Float16)aReg[p][j];
                *(h4*)(&As[nxt][(p * 32 + lrow) * LDK + lk]) = av;
            }
#pragma unroll
            for (int p = 0; p < 8; ++p) {
                h4 bv;
#pragma unroll
                for (int j = 0; j < 4; ++j)
                    bv[j] = (_Float16)__builtin_fmaf((float)bReg[p][j], sc[p], nzs[p]);
                *(h4*)(&Bs[nxt][(p * 32 + lrow) * LDK + lk]) = bv;
            }
            __syncthreads();
        }
    }

    // ---- epilogue: bias add + store (C layout: VGPR e -> M = h*8+e, N = lane%16) ----
#pragma unroll
    for (int fn = 0; fn < 4; ++fn) {
        const int n    = bn + waveN * 64 + fn * 16 + lr;
        const float bv = bias[n];
#pragma unroll
        for (int fm = 0; fm < 4; ++fm) {
            const int mbase = bm + waveM * 64 + fm * 16 + half * 8;
#pragma unroll
            for (int e = 0; e < 8; ++e) {
                out[(size_t)(mbase + e) * OUT_F + n] = acc[fm][fn][e] + bv;
            }
        }
    }
}

extern "C" void kernel_launch(void* const* d_in, const int* in_sizes, int n_in,
                              void* d_out, int out_size, void* d_ws, size_t ws_size,
                              hipStream_t stream) {
    const float* x  = (const float*)d_in[0];
    const int*   wq = (const int*)d_in[1];
    const float* ws = (const float*)d_in[2];
    const float* wz = (const float*)d_in[3];
    const float* bs = (const float*)d_in[4];
    float* out = (float*)d_out;

    dim3 grid(M_TOT / BM, OUT_F / BN);   // (64, 64): M fastest -> weight band reuse in L2
    dim3 block(256);
    qlinear_wmma_f16<<<grid, block, 0, stream>>>(x, wq, ws, wz, bs, out);
}
